// SoftCrossEntropyLoss_23124103921944
// MI455X (gfx1250) — compile-verified, matrix operand order checked by reading, not run on hardware
//
#include <hip/hip_runtime.h>
#include <hip/hip_bf16.h>

// SoftCrossEntropyLoss on MI455X (gfx1250).
// Memory-bound: 512 MB streamed once -> ~22us floor at 23.3 TB/s.
// Kernel 1: wave-per-row (one wave32 reads a full 512B row with a single
//           coalesced b128 load per array), NT cache hints (stream > 192MB L2),
//           butterfly wave reductions, deterministic block partials into d_ws.
// Kernel 2: single-wave WMMA reduction of the 2048 partials using the
//           codegen-confirmed v_wmma_f32_16x16x32_f16 against an all-ones B.

typedef float v4f  __attribute__((ext_vector_type(4)));
typedef __attribute__((ext_vector_type(16))) _Float16 v16h;
typedef __attribute__((ext_vector_type(8)))  float    v8f;

#define N_ROWS        524288
#define K_DIM         128
#define NUM_BLOCKS    2048
#define THREADS       256
#define WAVES         (THREADS / 32)                  // 8 waves / block
#define ROWS_PER_BLK  (N_ROWS / NUM_BLOCKS)           // 256
#define ROWS_PER_WAVE (ROWS_PER_BLK / WAVES)          // 32

__global__ void sce_partial_kernel(const float* __restrict__ input,
                                   const float* __restrict__ target,
                                   float* __restrict__ partials) {
    __shared__ float wsum[WAVES];
    const int lane = threadIdx.x & 31;
    const int wave = threadIdx.x >> 5;
    const int row0 = blockIdx.x * ROWS_PER_BLK + wave * ROWS_PER_WAVE;

    float acc = 0.0f;
    for (int i = 0; i < ROWS_PER_WAVE; ++i) {
        const int row = row0 + i;
        const v4f* ip = (const v4f*)(input  + (size_t)row * K_DIM);
        const v4f* tp = (const v4f*)(target + (size_t)row * K_DIM);

        // One wave -> 32 lanes x 16B = one full 512B row per load (perfectly
        // coalesced). Non-temporal: data is streamed once, never reused.
        v4f x = __builtin_nontemporal_load(ip + lane);
        v4f t = __builtin_nontemporal_load(tp + lane);

        // Prefetch next row's chunk (global_prefetch_b8 on gfx1250).
        if (i + 1 < ROWS_PER_WAVE) {
            __builtin_prefetch((const void*)(ip + (K_DIM / 4) + lane), 0, 0);
            __builtin_prefetch((const void*)(tp + (K_DIM / 4) + lane), 0, 0);
        }

        // Row max (numerical stabilization), wave32 butterfly.
        float m = fmaxf(fmaxf(x.x, x.y), fmaxf(x.z, x.w));
#pragma unroll
        for (int off = 16; off >= 1; off >>= 1)
            m = fmaxf(m, __shfl_xor(m, off, 32));

        // Per-lane partials: exp-sum, t.x dot, t sum.
        float e  = __expf(x.x - m) + __expf(x.y - m) +
                   __expf(x.z - m) + __expf(x.w - m);
        float d  = x.x * t.x + x.y * t.y + x.z * t.z + x.w * t.w;
        float ts = t.x + t.y + t.z + t.w;
#pragma unroll
        for (int off = 16; off >= 1; off >>= 1) {
            e  += __shfl_xor(e,  off, 32);
            d  += __shfl_xor(d,  off, 32);
            ts += __shfl_xor(ts, off, 32);
        }

        // loss = -sum t*(x - m - logS) = (m + logS)*sum(t) - sum(t*x)
        acc += (m + __logf(e)) * ts - d;
    }

    if (lane == 0) wsum[wave] = acc;          // acc is wave-uniform
    __syncthreads();
    if (threadIdx.x == 0) {
        float p = 0.0f;
#pragma unroll
        for (int w = 0; w < WAVES; ++w) p += wsum[w];
        partials[blockIdx.x] = p * (1.0f / (float)N_ROWS);  // pre-scale for mean
    }
}

// Sum NUM_BLOCKS fp32 partials with WMMA: D = A(16x32 f16) * ones(32x16) + C.
// The grand total of A's elements is invariant to how we pack them into the
// fragment, so any bijective (lane, j) -> element mapping is valid. Each
// lane's sum of its 8 C/D floats covers rows 0-7 (lanes 0-15) or 8-15
// (lanes 16-31) of D's column, so total = s + shfl_xor(s, 16).
__global__ void sce_reduce_wmma_kernel(const float* __restrict__ partials,
                                       float* __restrict__ out) {
    const int lane = threadIdx.x;   // launched with exactly 32 threads
    v8f acc = {};
    v16h ones;
#pragma unroll
    for (int j = 0; j < 16; ++j) ones[j] = (_Float16)1.0f;

    for (int c = 0; c < NUM_BLOCKS; c += 512) {   // 512 partials per WMMA
        v16h a;
#pragma unroll
        for (int j = 0; j < 16; ++j)
            a[j] = (_Float16)partials[c + lane * 16 + j];
        acc = __builtin_amdgcn_wmma_f32_16x16x32_f16(
            /*neg_a=*/false, a, /*neg_b=*/false, ones,
            /*c_mod=*/(short)0, acc, /*reuse_a=*/false, /*reuse_b=*/false);
    }

    float s = acc[0] + acc[1] + acc[2] + acc[3] +
              acc[4] + acc[5] + acc[6] + acc[7];
    s += __shfl_xor(s, 16, 32);
    if (lane == 0) out[0] = s;
}

extern "C" void kernel_launch(void* const* d_in, const int* in_sizes, int n_in,
                              void* d_out, int out_size, void* d_ws, size_t ws_size,
                              hipStream_t stream) {
    const float* input  = (const float*)d_in[0];
    const float* target = (const float*)d_in[1];
    float* partials = (float*)d_ws;              // NUM_BLOCKS floats = 8 KB
    float* out      = (float*)d_out;

    sce_partial_kernel<<<NUM_BLOCKS, THREADS, 0, stream>>>(input, target, partials);
    sce_reduce_wmma_kernel<<<1, 32, 0, stream>>>(partials, out);
}
